// HFLSTM_21010980012084
// MI455X (gfx1250) — matrix operand order repaired and checked by reading, not compiled, not run on hardware
//
#include <hip/hip_runtime.h>
#include <stdint.h>

// ---------------------------------------------------------------------------
// HFLSTM on MI455X (gfx1250, wave32, WMMA).
//
// 8 persistent workgroups x 64 batch rows; 389-step recurrence with only
// intra-workgroup barriers. 512 threads/WG (16 waves, 4/SIMD, ~256 VGPRs/wave):
//   - wave wv owns hidden tile ht=wv and ALL 4 M-tiles x 4 gates
//     (acc = 128 VGPRs; each B-fragment feeds 4 WMMAs, A-fragment feeds 4)
//   - c-state lives in VGPRs; i/f/g/o update is pure-register
//   x0  = relu(x[64,16] @ W_in.T)  via WMMA (K padded to 32, weights in LDS)
//   g   = [x0|h][64,512] @ [W_ih;W_hh].T  -> 256 bf16 WMMA / wave / step
//   prev = h @ W_out.T  (N=1) via lane-reduced dot
// Gate weights (1MB bf16) stream from L2 once per WG per step; next-step x is
// prefetched into registers during the gate GEMM. Per-lane LDS addresses in
// the (a)/(b) sections are recomputed per step (asm remat seed) instead of
// being LICM-hoisted and spilled to scratch.
// ---------------------------------------------------------------------------

#define RHO_T  365
#define HOR_T  24
#define TOT_T  (RHO_T + HOR_T)
#define BATCH  512
#define HID    256
#define FIN    15
#define NGATE  1024          // 4*H
#define KTOT   512           // x0(256) + h(256)
#define BSLICE 64            // batch rows per workgroup
#define NWG    (BATCH / BSLICE)
#define NTHR   512           // 16 waves

typedef __attribute__((ext_vector_type(16))) __bf16 v16bf;
typedef __attribute__((ext_vector_type(8)))  float  v8f;
typedef __attribute__((ext_vector_type(4)))  unsigned int u32x4;

union Frag { u32x4 u[2]; v16bf v; };   // 32B: one WMMA A/B operand (16 bf16)

__device__ __forceinline__ unsigned short f2bf(float x) {
  union { float f; unsigned u; } c; c.f = x;
  unsigned u = c.u + 0x7FFFu + ((c.u >> 16) & 1u);   // round-to-nearest-even
  return (unsigned short)(u >> 16);
}
__device__ __forceinline__ float bf2f(unsigned short b) {
  union { float f; unsigned u; } c; c.u = ((unsigned)b) << 16; return c.f;
}
__device__ __forceinline__ float sigf(float x) { return 1.f / (1.f + __expf(-x)); }
__device__ __forceinline__ float tanh_fast(float x) {
  float t = __expf(-2.f * fabsf(x));
  float r = (1.f - t) / (1.f + t);
  return copysignf(r, x);
}

__device__ __forceinline__ v8f wmma_bf16(const Frag& a, const Frag& b, v8f c) {
  return __builtin_amdgcn_wmma_f32_16x16x32_bf16(
      false, a.v, false, b.v, (short)0, c, false, false);
}

// --------------------------- weight prep -----------------------------------
// Wt[n][k]  : bf16 [1024][512], k<256 -> W_ih[n][k], else W_hh[n][k-256]
// WinT[j][k]: bf16 [256][32],   k<16  -> W_in[j][k], else 0 (K pad for WMMA)
// biasC[n]  : f32 b_ih+b_hh
__global__ void hflstm_prep(const float* __restrict__ W_ih,
                            const float* __restrict__ W_hh,
                            const float* __restrict__ b_ih,
                            const float* __restrict__ b_hh,
                            const float* __restrict__ W_in,
                            unsigned short* __restrict__ Wt,
                            float* __restrict__ biasC,
                            unsigned short* __restrict__ WinT) {
  int idx = blockIdx.x * 256 + threadIdx.x;
  if (idx < NGATE * KTOT) {
    int nr = idx >> 9, k = idx & 511;
    float v = (k < HID) ? W_ih[nr * HID + k] : W_hh[nr * HID + (k - HID)];
    Wt[idx] = f2bf(v);
  }
  if (idx < NGATE) biasC[idx] = b_ih[idx] + b_hh[idx];
  if (idx < HID * 32) {
    int j = idx >> 5, k = idx & 31;
    WinT[idx] = (k < 16) ? f2bf(W_in[j * 16 + k]) : (unsigned short)0;
  }
}

// --------------------------- LDS layout ------------------------------------
#define OFF_X0   0                       // ushort [64][256]  bf16 x0
#define OFF_H0   32768                   // ushort [64][256]  bf16 h ping
#define OFF_H1   65536                   // ushort [64][256]  bf16 h pong
#define OFF_XB   98304                   // ushort [64][32]   bf16 x (K-padded)
#define OFF_WINT 102400                  // ushort [256][32]  bf16 W_in (padded)
#define OFF_BIN  118784                  // float  [256]
#define OFF_WOUT 119808                  // float  [256]
#define OFF_PREV 120832                  // float  [64]
#define SMEM_BYTES 121088

// Prefetch next step's x element (cc==15 in hor phase is patched from prevS
// at store time, so return 0 there).
__device__ __forceinline__ float load_x(const float* __restrict__ xfc_rho,
                                        const float* __restrict__ xfc_hor,
                                        const float* __restrict__ xq_rho,
                                        int s, int b0, int i) {
  int r = i >> 4, cc = i & 15;
  if (cc < FIN) {
    return (s < RHO_T)
        ? xfc_rho[((size_t)s * BATCH + b0 + r) * FIN + cc]
        : xfc_hor[((size_t)(s - RHO_T) * BATCH + b0 + r) * FIN + cc];
  }
  return (s < RHO_T) ? xq_rho[(size_t)s * BATCH + b0 + r] : 0.f;
}

// --------------------------- main kernel -----------------------------------
__global__ __launch_bounds__(NTHR, 1)
void hflstm_persist(const float* __restrict__ xfc_rho,
                    const float* __restrict__ xfc_hor,
                    const float* __restrict__ xq_rho,
                    const float* __restrict__ b_in,
                    const float* __restrict__ W_out,
                    const float* __restrict__ b_out,
                    const unsigned short* __restrict__ Wt,
                    const float* __restrict__ biasC,
                    const unsigned short* __restrict__ WinT,
                    float* __restrict__ out) {
  extern __shared__ char smem[];
  unsigned short* x0buf = (unsigned short*)(smem + OFF_X0);
  unsigned short* hbufA = (unsigned short*)(smem + OFF_H0);
  unsigned short* hbufB = (unsigned short*)(smem + OFF_H1);
  unsigned short* xb16  = (unsigned short*)(smem + OFF_XB);
  unsigned short* winS  = (unsigned short*)(smem + OFF_WINT);
  float* binS  = (float*)(smem + OFF_BIN);
  float* WoutS = (float*)(smem + OFF_WOUT);
  float* prevS = (float*)(smem + OFF_PREV);

  const int tid = threadIdx.x;
  const int b0  = blockIdx.x * BSLICE;

  if (tid < HID) { binS[tid] = b_in[tid]; WoutS[tid] = W_out[tid]; }
  for (int i = tid; i < HID * 32; i += NTHR) winS[i] = WinT[i];
  for (int i = tid; i < BSLICE * HID; i += NTHR) { hbufA[i] = 0; hbufB[i] = 0; }

  // Gate GEMM tiling: 16 waves; wave wv owns hidden tile ht=wv, all 4 M-tiles.
  const int lane = tid & 31;
  const int wv   = tid >> 5;        // 0..15 == hidden tile
  const int g    = lane >> 4;       // lane group (K-half select)
  const int n    = lane & 15;       // row within A-tile / col within B-tile

  const float bI   = biasC[0 * HID + wv * 16 + n];
  const float bF   = biasC[1 * HID + wv * 16 + n];
  const float bG   = biasC[2 * HID + wv * 16 + n];
  const float bO4  = biasC[3 * HID + wv * 16 + n];
  const float bOut = b_out[0];

  // Per-wave weight base: row = wv*16+n inside each gate block of Wt.
  const unsigned short* wbase = Wt + (size_t)(wv * 16 + n) * KTOT + g * 16;

  const v8f z8 = {0.f, 0.f, 0.f, 0.f, 0.f, 0.f, 0.f, 0.f};
  v8f cst[4] = { z8, z8, z8, z8 };  // persistent cell state (in VGPRs)

  // Prime the x pipeline for step 0.
  float xv[2];
  #pragma unroll
  for (int s2 = 0; s2 < 2; ++s2)
    xv[s2] = load_x(xfc_rho, xfc_hor, xq_rho, 0, b0, tid + s2 * NTHR);
  __syncthreads();

  int cur = 0;
  for (int step = 0; step < TOT_T; ++step) {
    const bool rho = step < RHO_T;
    const int  th  = step - RHO_T;
    unsigned short* hcur = cur ? hbufB : hbufA;
    unsigned short* hnxt = cur ? hbufA : hbufB;

    // Remat seed: derive (a)/(b) per-lane indices from an opaque copy of tid
    // so they are recomputed each step (cheap VALU) rather than hoisted out
    // of the 389-step loop and spilled to scratch by regalloc.
    int tidv = tid;
    asm volatile("" : "+v"(tidv));
    const int laneR = tidv & 31;
    const int wvR   = tidv >> 5;
    const int gR    = laneR >> 4;
    const int nR    = laneR & 15;
    const int mtXR  = wvR >> 2;          // x0 tiling: M-tile
    const int ntX0R = (wvR & 3) * 4;     // x0 tiling: first of 4 N-tiles

    // (a) commit the prefetched x_t [64,16] -> bf16 LDS (K-padded to 32);
    //     hor phase: col 15 = fed-back prev (from prevS, written last step)
    #pragma unroll
    for (int s2 = 0; s2 < 2; ++s2) {
      int i = tidv + s2 * NTHR;
      int r = i >> 4, cc = i & 15;
      float v = xv[s2];
      if (cc == FIN && !rho) v = prevS[r];
      xb16[r * 32 + cc]      = f2bf(v);
      xb16[r * 32 + 16 + cc] = 0;
    }
    __syncthreads();

    // (b) x0 = relu(x @ W_in.T + b_in) via WMMA (4 tiles/wave, weights in LDS)
    {
      const unsigned short* ap = xb16 + (mtXR * 16 + nR) * 32 + gR * 16;
      Frag a;
      a.u[0] = *(const u32x4*)(ap);
      a.u[1] = *(const u32x4*)(ap + 8);
      #pragma unroll
      for (int i = 0; i < 4; ++i) {
        int nt = ntX0R + i;
        const unsigned short* wp = winS + (nt * 16 + nR) * 32 + gR * 16;
        Frag b;
        b.u[0] = *(const u32x4*)(wp);
        b.u[1] = *(const u32x4*)(wp + 8);
        v8f c = wmma_bf16(a, b, z8);
        float bb = binS[nt * 16 + nR];
        #pragma unroll
        for (int e = 0; e < 8; ++e) {
          float v = c[e] + bb;
          v = v > 0.f ? v : 0.f;
          int row = mtXR * 16 + e + 8 * gR;
          x0buf[row * HID + nt * 16 + nR] = f2bf(v);
        }
      }
    }

    // Prefetch next step's x while the gate GEMM runs (loads retire with the
    // first B-fragment wait; values consumed only at next iteration's (a)).
    if (step + 1 < TOT_T) {
      #pragma unroll
      for (int s2 = 0; s2 < 2; ++s2)
        xv[s2] = load_x(xfc_rho, xfc_hor, xq_rho, step + 1, b0, tid + s2 * NTHR);
    }
    __syncthreads();

    // (c) gates = [x0|h] @ Wt.T  (bf16 WMMA, f32 acc; 16 WMMA per k-step)
    v8f acc[4][4];
    #pragma unroll
    for (int mt = 0; mt < 4; ++mt)
      #pragma unroll
      for (int gt = 0; gt < 4; ++gt) acc[mt][gt] = z8;

    #pragma unroll 2
    for (int k0 = 0; k0 < KTOT; k0 += 32) {
      const unsigned short* Asrc = (k0 < HID) ? x0buf : hcur;
      const int kk = (k0 < HID) ? k0 : (k0 - HID);
      Frag afr[4];
      #pragma unroll
      for (int mt = 0; mt < 4; ++mt) {
        const unsigned short* p =
            Asrc + ((mt * 16 + n) * HID) + kk + g * 16;
        afr[mt].u[0] = *(const u32x4*)(p);        // K+0..7  / +16..23
        afr[mt].u[1] = *(const u32x4*)(p + 8);    // K+8..15 / +24..31
      }
      // B fragments pipelined 2-deep: <=2 in flight, 16 regs.
      Frag bfr[4];
      {
        const unsigned short* wp = wbase + 0 * (HID * KTOT) + k0;
        bfr[0].u[0] = *(const u32x4*)(wp);
        bfr[0].u[1] = *(const u32x4*)(wp + 8);
        wp = wbase + 1 * (size_t)(HID * KTOT) + k0;
        bfr[1].u[0] = *(const u32x4*)(wp);
        bfr[1].u[1] = *(const u32x4*)(wp + 8);
      }
      #pragma unroll
      for (int gt = 0; gt < 4; ++gt) {
        if (gt + 2 < 4) {
          const unsigned short* wp =
              wbase + (size_t)(gt + 2) * (HID * KTOT) + k0;
          bfr[gt + 2].u[0] = *(const u32x4*)(wp);
          bfr[gt + 2].u[1] = *(const u32x4*)(wp + 8);
        }
        acc[0][gt] = wmma_bf16(afr[0], bfr[gt], acc[0][gt]);
        acc[1][gt] = wmma_bf16(afr[1], bfr[gt], acc[1][gt]);
        acc[2][gt] = wmma_bf16(afr[2], bfr[gt], acc[2][gt]);
        acc[3][gt] = wmma_bf16(afr[3], bfr[gt], acc[3][gt]);
      }
    }

    // (d) LSTM cell update, pure-register (i,f,g,o co-located per wave)
    #pragma unroll
    for (int mt = 0; mt < 4; ++mt) {
      v8f gi = acc[mt][0], gf = acc[mt][1], gg = acc[mt][2], go = acc[mt][3];
      #pragma unroll
      for (int e = 0; e < 8; ++e) {
        float iv = gi[e] + bI, fv = gf[e] + bF;
        float gv = gg[e] + bG, ov = go[e] + bO4;
        float cnew = sigf(fv) * cst[mt][e] + sigf(iv) * tanh_fast(gv);
        float hnew = sigf(ov) * tanh_fast(cnew);
        cst[mt][e] = cnew;
        int row = mt * 16 + e + 8 * g;              // C-tile M = v + 8*laneGrp
        hnxt[row * HID + wv * 16 + n] = f2bf(hnew);
      }
    }
    __syncthreads();

    // (e) prev = h @ W_out.T + b_out (N=1): 4 lanes per row + shfl reduce
    if (tid < 4 * BSLICE) {
      int r = tid >> 2, part = tid & 3;
      const unsigned short* hp = hnxt + r * HID + part * 64;
      const float* wo = WoutS + part * 64;
      float s = 0.f;
      #pragma unroll 8
      for (int k = 0; k < 64; ++k) s += bf2f(hp[k]) * wo[k];
      s += __shfl_xor(s, 1);
      s += __shfl_xor(s, 2);
      if (part == 0) {
        s += bOut;
        prevS[r] = s;
        if (!rho) out[(size_t)th * BATCH + b0 + r] = s;
      }
    }
    cur ^= 1;
    __syncthreads();
  }
}

// --------------------------- launch ----------------------------------------
extern "C" void kernel_launch(void* const* d_in, const int* in_sizes, int n_in,
                              void* d_out, int out_size, void* d_ws, size_t ws_size,
                              hipStream_t stream) {
  (void)in_sizes; (void)n_in; (void)out_size; (void)ws_size;
  const float* xfc_rho = (const float*)d_in[0];
  const float* xfc_hor = (const float*)d_in[1];
  const float* xq_rho  = (const float*)d_in[2];
  /* d_in[3] = xq_hor: unused (teacher_forcing_ratio == 0) */
  const float* W_in  = (const float*)d_in[4];
  const float* b_in  = (const float*)d_in[5];
  const float* W_ih  = (const float*)d_in[6];
  const float* W_hh  = (const float*)d_in[7];
  const float* b_ih  = (const float*)d_in[8];
  const float* b_hh  = (const float*)d_in[9];
  const float* W_out = (const float*)d_in[10];
  const float* b_out = (const float*)d_in[11];

  unsigned short* Wt    = (unsigned short*)d_ws;                     // 1 MB
  float*          biasC = (float*)((char*)d_ws + (size_t)NGATE * KTOT * 2);
  unsigned short* WinT  = (unsigned short*)((char*)biasC + NGATE * sizeof(float));

  hflstm_prep<<<(NGATE * KTOT + 255) / 256, 256, 0, stream>>>(
      W_ih, W_hh, b_ih, b_hh, W_in, Wt, biasC, WinT);

  (void)hipFuncSetAttribute((const void*)hflstm_persist,
                            hipFuncAttributeMaxDynamicSharedMemorySize,
                            SMEM_BYTES);
  hflstm_persist<<<NWG, NTHR, SMEM_BYTES, stream>>>(
      xfc_rho, xfc_hor, xq_rho, b_in, W_out, b_out, Wt, biasC, WinT,
      (float*)d_out);
}
